// UpFlow_1786706395669
// MI455X (gfx1250) — matrix operand order repaired and checked by reading, not compiled, not run on hardware
//
#include <hip/hip_runtime.h>

// ---------------------------------------------------------------------------
// UpFlow refinement net for gfx1250 (MI455X):
//   warp -> cost volume(81) -> concat(211->pad224, bf16)
//   6x [depthwise3x3 -> pointwise GEMM via v_wmma_f32_16x16x32_bf16 -> mish]
// All intermediates bf16; WMMA f32 accumulate; final layer f32 to d_out.
// ---------------------------------------------------------------------------

typedef __attribute__((ext_vector_type(16))) __bf16 v16bf;
typedef __attribute__((ext_vector_type(8)))  float  v8f;

#define DEV static __device__ __forceinline__

static constexpr int  PB  = 8;     // batch
static constexpr int  PH  = 128;   // height
static constexpr int  PW  = 128;   // width
static constexpr int  PC  = 128;   // feature channels
static constexpr long PIX = (long)PB * PH * PW;   // 131072 pixels

DEV unsigned short f2bf(float f) {
    unsigned int u = __float_as_uint(f);
    unsigned int r = u + 0x7FFFu + ((u >> 16) & 1u);   // round-nearest-even
    return (unsigned short)(r >> 16);
}
DEV float bf2f(unsigned short s) {
    return __uint_as_float(((unsigned int)s) << 16);
}
DEV float mishf(float v) {
    float sp = (v > 20.0f) ? v : log1pf(expf(v));      // softplus, stable
    return v * tanhf(sp);
}

// ---------------------------------------------------------------------------
// Kernel 1: bilinear warp of nxt by flo (bf16 out), plus write prv/flo/pad
// channels of the concat buffer x0 (224 ch).  One thread = (pixel, 4 chans).
// ---------------------------------------------------------------------------
__global__ __launch_bounds__(256)
void warp_kernel(const float* __restrict__ prv, const float* __restrict__ nxt,
                 const float* __restrict__ flo,
                 unsigned short* __restrict__ nxtw,   // [PIX,128] bf16
                 unsigned short* __restrict__ x0)     // [PIX,224] bf16
{
    unsigned idx = blockIdx.x * 256u + threadIdx.x;   // PIX*32 threads
    int p  = (int)(idx >> 5);
    int t  = (int)(idx & 31u);
    int c0 = t * 4;
    int b = p >> 14, hw = p & 16383, h = hw >> 7, w = hw & 127;

    float fx = flo[2 * (size_t)p], fy = flo[2 * (size_t)p + 1];
    float xx = (float)w + fx, yy = (float)h + fy;
    int x0i = (int)xx, y0i = (int)yy;                 // trunc toward zero
    int x1i = x0i + 1, y1i = y0i + 1;
    int x0c = min(max(x0i, 0), PW - 1), x1c = min(max(x1i, 0), PW - 1);
    int y0c = min(max(y0i, 0), PH - 1), y1c = min(max(y1i, 0), PH - 1);
    float x0f = (float)x0c, x1f = (float)x1c, y0f = (float)y0c, y1f = (float)y1c;
    float wa = (x1f - xx) * (y1f - yy);
    float wb = (x1f - xx) * (yy - y0f);
    float wc = (xx - x0f) * (y1f - yy);
    float wd = (xx - x0f) * (yy - y0f);

    size_t base = (size_t)(b << 14);
    const float4* Ia = (const float4*)(nxt + (((base + (size_t)(y0c << 7) + x0c) << 7) + c0));
    const float4* Ib = (const float4*)(nxt + (((base + (size_t)(y1c << 7) + x0c) << 7) + c0));
    const float4* Ic = (const float4*)(nxt + (((base + (size_t)(y0c << 7) + x1c) << 7) + c0));
    const float4* Id = (const float4*)(nxt + (((base + (size_t)(y1c << 7) + x1c) << 7) + c0));
    float4 A = *Ia, B = *Ib, C = *Ic, D = *Id;
    float r0 = wa * A.x + wb * B.x + wc * C.x + wd * D.x;
    float r1 = wa * A.y + wb * B.y + wc * C.y + wd * D.y;
    float r2 = wa * A.z + wb * B.z + wc * C.z + wd * D.z;
    float r3 = wa * A.w + wb * B.w + wc * C.w + wd * D.w;

    uint2 pk;
    pk.x = (unsigned)f2bf(r0) | ((unsigned)f2bf(r1) << 16);
    pk.y = (unsigned)f2bf(r2) | ((unsigned)f2bf(r3) << 16);
    *(uint2*)(nxtw + (size_t)p * PC + c0) = pk;

    // concat: prv at channels [81,209)
    float4 pv = *(const float4*)(prv + (size_t)p * PC + c0);
    unsigned short* xrow = x0 + (size_t)p * 224;
    xrow[81 + c0 + 0] = f2bf(pv.x);
    xrow[81 + c0 + 1] = f2bf(pv.y);
    xrow[81 + c0 + 2] = f2bf(pv.z);
    xrow[81 + c0 + 3] = f2bf(pv.w);
    if (t == 0) {
        xrow[209] = f2bf(fx);
        xrow[210] = f2bf(fy);
        #pragma unroll
        for (int i = 211; i < 224; ++i) xrow[i] = 0;   // zero pad channels
    }
}

// ---------------------------------------------------------------------------
// Kernel 2: cost volume.  One wave per pixel; lane owns offsets
// k = lane, lane+32, lane+64 (k<81) -> no cross-lane reduction needed.
// prv loads are wave-uniform; neighbor rows served from L2 (192MB).
// ---------------------------------------------------------------------------
__global__ __launch_bounds__(256)
void cost_kernel(const float* __restrict__ prv,
                 const unsigned short* __restrict__ nxtw,
                 unsigned short* __restrict__ x0)
{
    int tid = threadIdx.x, lane = tid & 31, wid = tid >> 5;
    int p = blockIdx.x * 8 + wid;
    int b = p >> 14, hw = p & 16383, h = hw >> 7, w = hw & 127;

    float acc[3] = {0.f, 0.f, 0.f};
    size_t q[3];
    bool val[3];
    #pragma unroll
    for (int s = 0; s < 3; ++s) {
        int k = lane + 32 * s;
        int di = k / 9 - 4, dj = k % 9 - 4;
        int hy = h + di, wx = w + dj;
        val[s] = (k < 81) && ((unsigned)hy < (unsigned)PH) && ((unsigned)wx < (unsigned)PW);
        q[s] = ((size_t)(b << 14) + (size_t)(hy << 7) + (size_t)wx) << 7;
        if (!val[s]) q[s] = 0;
    }

    const float4* prow = (const float4*)(prv + (size_t)p * PC);
    #pragma unroll 4
    for (int i = 0; i < 32; ++i) {
        float4 pv = prow[i];
        #pragma unroll
        for (int s = 0; s < 3; ++s) {
            if (val[s]) {
                uint2 u = *(const uint2*)(nxtw + q[s] + i * 4);
                float n0 = bf2f((unsigned short)(u.x & 0xFFFF));
                float n1 = bf2f((unsigned short)(u.x >> 16));
                float n2 = bf2f((unsigned short)(u.y & 0xFFFF));
                float n3 = bf2f((unsigned short)(u.y >> 16));
                acc[s] += pv.x * n0 + pv.y * n1 + pv.z * n2 + pv.w * n3;
            }
        }
    }
    unsigned short* xrow = x0 + (size_t)p * 224;
    #pragma unroll
    for (int s = 0; s < 3; ++s) {
        int k = lane + 32 * s;
        if (k < 81) xrow[k] = f2bf(acc[s] * 0.0078125f);   // mean over 128
    }
}

// ---------------------------------------------------------------------------
// Kernel 3: depthwise 3x3 (SAME).  Thread = (pixel, 4 channels), bf16 io.
// ---------------------------------------------------------------------------
template <int CP, int CIN>
__global__ __launch_bounds__(256)
void dw_kernel(const unsigned short* __restrict__ xin,
               const float* __restrict__ dwt,      // [9, CIN]
               unsigned short* __restrict__ yout)
{
    unsigned idx = blockIdx.x * 256u + threadIdx.x;   // PIX*CP/4 threads
    int p  = (int)(idx / (CP / 4));
    int c0 = (int)(idx % (CP / 4)) * 4;
    int b = p >> 14, hw = p & 16383, h = hw >> 7, w = hw & 127;

    float wgt[9][4];
    #pragma unroll
    for (int kk = 0; kk < 9; ++kk)
        #pragma unroll
        for (int e = 0; e < 4; ++e) {
            int c = c0 + e;
            wgt[kk][e] = (c < CIN) ? dwt[kk * CIN + c] : 0.f;
        }

    float acc[4] = {0.f, 0.f, 0.f, 0.f};
    #pragma unroll
    for (int ky = 0; ky < 3; ++ky) {
        int hy = h + ky - 1;
        if ((unsigned)hy >= (unsigned)PH) continue;
        #pragma unroll
        for (int kx = 0; kx < 3; ++kx) {
            int wx = w + kx - 1;
            if ((unsigned)wx >= (unsigned)PW) continue;
            size_t qb = ((size_t)(b << 14) + (size_t)(hy << 7) + (size_t)wx) * CP + c0;
            uint2 u = *(const uint2*)(xin + qb);
            float v0 = bf2f((unsigned short)(u.x & 0xFFFF));
            float v1 = bf2f((unsigned short)(u.x >> 16));
            float v2 = bf2f((unsigned short)(u.y & 0xFFFF));
            float v3 = bf2f((unsigned short)(u.y >> 16));
            int kk = ky * 3 + kx;
            acc[0] += wgt[kk][0] * v0;
            acc[1] += wgt[kk][1] * v1;
            acc[2] += wgt[kk][2] * v2;
            acc[3] += wgt[kk][3] * v3;
        }
    }
    uint2 pk;
    pk.x = (unsigned)f2bf(acc[0]) | ((unsigned)f2bf(acc[1]) << 16);
    pk.y = (unsigned)f2bf(acc[2]) | ((unsigned)f2bf(acc[3]) << 16);
    *(uint2*)(yout + (size_t)p * CP + c0) = pk;
}

// ---------------------------------------------------------------------------
// Kernel 4: pointwise 1x1 conv == GEMM [M=PIX, K=KP, N=NP] via WMMA bf16.
// 8 waves/block; wave owns a 16-row M tile and all NJ column tiles.
// Weights pre-swizzled into LDS in exact B-fragment order.
// ISA layouts (CDNA5 ISA):
//   A 16x32 bf16: lane(m=l&15, hi=l>>4): slots0-7 K=hi*8+i, slots8-15 K=16+hi*8+i
//   B 32x16 bf16: lane(n=l&15, hi=l>>4): slot i K=hi*16+i (order this code uses for A and B)
//   D 16x16 f32 : acc[v] -> row m=v+hi*8, col n=l&15
// ---------------------------------------------------------------------------
DEV v16bf frag_from(uint4 q0, uint4 q1) {
    union { unsigned int u[8]; v16bf v; } f;
    f.u[0] = q0.x; f.u[1] = q0.y; f.u[2] = q0.z; f.u[3] = q0.w;
    f.u[4] = q1.x; f.u[5] = q1.y; f.u[6] = q1.z; f.u[7] = q1.w;
    return f.v;
}

template <int KP, int NP, int CIN, int COUT, bool ACT, bool F32OUT>
__global__ __launch_bounds__(256)
void pw_kernel(const unsigned short* __restrict__ yin,   // [PIX,KP] bf16
               const float* __restrict__ pwt,            // [CIN,COUT]
               const float* __restrict__ bias,           // [COUT] or null
               unsigned short* __restrict__ xout,        // [PIX,NP] bf16
               float* __restrict__ fout)                 // [PIX,COUT] f32
{
    constexpr int NT = KP / 32;   // K-steps
    constexpr int NJ = NP / 16;   // N tiles
    __shared__ uint4 ldsW[NT * NJ * 32 * 2];
    __shared__ float ldsB[NP];

    int tid = threadIdx.x;

    // ---- stage weights into LDS, pre-swizzled into B-fragment order ----
    for (int idx = tid; idx < NT * NJ * 32; idx += 256) {
        int l  = idx & 31;
        int tj = idx >> 5;
        int j  = tj % NJ;
        int t  = tj / NJ;
        int n     = j * 16 + (l & 15);
        int kbase = t * 32 + (l >> 4) * 16;
        unsigned int u[8];
        #pragma unroll
        for (int i = 0; i < 16; i += 2) {
            float a0 = 0.f, a1 = 0.f;
            int k0 = kbase + i, k1 = kbase + i + 1;
            if (n < COUT) {
                if (k0 < CIN) a0 = pwt[(size_t)k0 * COUT + n];
                if (k1 < CIN) a1 = pwt[(size_t)k1 * COUT + n];
            }
            u[i >> 1] = (unsigned)f2bf(a0) | ((unsigned)f2bf(a1) << 16);
        }
        ldsW[idx * 2 + 0] = make_uint4(u[0], u[1], u[2], u[3]);
        ldsW[idx * 2 + 1] = make_uint4(u[4], u[5], u[6], u[7]);
    }
    if (tid < NP) {
        float bv = 0.f;
        if (ACT && tid < COUT) bv = bias[tid];
        ldsB[tid] = bv;
    }
    __syncthreads();

    int lane = tid & 31, wid = tid >> 5;
    long m0  = (long)blockIdx.x * 128 + wid * 16;
    int mrow = lane & 15, hi = lane >> 4;
    const unsigned short* arow = yin + (size_t)(m0 + mrow) * KP + hi * 16;

    v8f acc[NJ] = {};
    #pragma unroll
    for (int t = 0; t < NT; ++t) {
        const uint4* ap = (const uint4*)(arow + t * 32);
        v16bf a = frag_from(ap[0], ap[1]);   // K = hi*16+0..15
        #pragma unroll
        for (int j = 0; j < NJ; ++j) {
            const uint4* bp = &ldsW[((t * NJ + j) * 32 + lane) * 2];
            v16bf b = frag_from(bp[0], bp[1]);
            acc[j] = __builtin_amdgcn_wmma_f32_16x16x32_bf16(
                false, a, false, b, (short)0, acc[j], false, false);
        }
    }

    // ---- epilogue: bias + mish, store bf16 (or f32 for final layer) ----
    int n0 = lane & 15;
    #pragma unroll
    for (int j = 0; j < NJ; ++j) {
        int n = j * 16 + n0;
        float bia = ldsB[n];
        #pragma unroll
        for (int v = 0; v < 8; ++v) {
            long m = m0 + v + hi * 8;
            float valf = acc[j][v] + bia;
            if (ACT) valf = mishf(valf);
            if (F32OUT) {
                if (n < COUT) fout[(size_t)m * COUT + n] = valf;
            } else {
                xout[(size_t)m * NP + n] = f2bf(valf);
            }
        }
    }
}

// ---------------------------------------------------------------------------
// Host-side launch.  Workspace layout (needs ~151 MB):
//   regionX0 : PIX*224 bf16  (x0 concat; later reused as activation buffer)
//   regionY  : PIX*224 bf16  (warped image first, then depthwise outputs)
//   regionA  : PIX*128 bf16  (activation ping buffer)
// ---------------------------------------------------------------------------
extern "C" void kernel_launch(void* const* d_in, const int* in_sizes, int n_in,
                              void* d_out, int out_size, void* d_ws, size_t ws_size,
                              hipStream_t stream) {
    const float* prv = (const float*)d_in[0];
    const float* nxt = (const float*)d_in[1];
    const float* flo = (const float*)d_in[2];
    const float* dwt[6] = {(const float*)d_in[3],  (const float*)d_in[6],
                           (const float*)d_in[9],  (const float*)d_in[12],
                           (const float*)d_in[15], (const float*)d_in[18]};
    const float* pwt[6] = {(const float*)d_in[4],  (const float*)d_in[7],
                           (const float*)d_in[10], (const float*)d_in[13],
                           (const float*)d_in[16], (const float*)d_in[19]};
    const float* bia[5] = {(const float*)d_in[5],  (const float*)d_in[8],
                           (const float*)d_in[11], (const float*)d_in[14],
                           (const float*)d_in[17]};
    float* out = (float*)d_out;

    char* ws = (char*)d_ws;
    unsigned short* X0 = (unsigned short*)ws;                          // PIX*224
    unsigned short* Y  = (unsigned short*)(ws + (size_t)PIX * 224 * 2);
    unsigned short* A  = (unsigned short*)(ws + (size_t)PIX * 224 * 4); // PIX*128
    unsigned short* NXTW = Y;   // warped image aliases Y (dead before dw0)

    const int nP = (int)PIX;
    warp_kernel<<<nP * 32 / 256, 256, 0, stream>>>(prv, nxt, flo, NXTW, X0);
    cost_kernel<<<nP / 8, 256, 0, stream>>>(prv, NXTW, X0);

    // L0: 224(211) -> 128
    dw_kernel<224, 211><<<nP * 56 / 256, 256, 0, stream>>>(X0, dwt[0], Y);
    pw_kernel<224, 128, 211, 128, true, false>
        <<<nP / 128, 256, 0, stream>>>(Y, pwt[0], bia[0], A, nullptr);
    // L1: 128 -> 128
    dw_kernel<128, 128><<<nP * 32 / 256, 256, 0, stream>>>(A, dwt[1], Y);
    pw_kernel<128, 128, 128, 128, true, false>
        <<<nP / 128, 256, 0, stream>>>(Y, pwt[1], bia[1], X0, nullptr);
    // L2: 128 -> 96
    dw_kernel<128, 128><<<nP * 32 / 256, 256, 0, stream>>>(X0, dwt[2], Y);
    pw_kernel<128, 96, 128, 96, true, false>
        <<<nP / 128, 256, 0, stream>>>(Y, pwt[2], bia[2], A, nullptr);
    // L3: 96 -> 64
    dw_kernel<96, 96><<<nP * 24 / 256, 256, 0, stream>>>(A, dwt[3], Y);
    pw_kernel<96, 64, 96, 64, true, false>
        <<<nP / 128, 256, 0, stream>>>(Y, pwt[3], bia[3], X0, nullptr);
    // L4: 64 -> 32
    dw_kernel<64, 64><<<nP * 16 / 256, 256, 0, stream>>>(X0, dwt[4], Y);
    pw_kernel<64, 32, 64, 32, true, false>
        <<<nP / 128, 256, 0, stream>>>(Y, pwt[4], bia[4], A, nullptr);
    // L5: 32 -> 2 (f32, no bias/act)
    dw_kernel<32, 32><<<nP * 8 / 256, 256, 0, stream>>>(A, dwt[5], Y);
    pw_kernel<32, 16, 32, 2, false, true>
        <<<nP / 128, 256, 0, stream>>>(Y, pwt[5], nullptr, nullptr, out);
}
